// dgl_gat_18047452578198
// MI455X (gfx1250) — matrix operand order repaired and checked by reading, not compiled, not run on hardware
//
#include <hip/hip_runtime.h>
#include <hip/hip_bf16.h>
#include <math.h>

// ---------------------------------------------------------------------------
// 2-layer GAT (DGL GATConv semantics) for MI455X / gfx1250.
//   N=50000 nodes, E=800000 edges, IN=256, H1=4 x D=64, H2=1 x C=40.
// GEMMs: v_wmma_f32_16x16x32_bf16, fp32 accumulate, bf16 hi/lo split
//   (Ahi*Bhi + Ahi*Blo + Alo*Bhi) for near-fp32 accuracy.
//   A pre-split to row-major bf16 hi/lo; B pre-packed fragment-major
//   (zero-padded). Each wave computes a 32x16 C block: B fragments are
//   reused by 6 independent WMMAs per K-step (latency hiding + half the
//   B traffic).
// Edge softmax + aggregation: f32 atomics, L2-resident accumulators.
// ---------------------------------------------------------------------------

#define N_NODES 50000
#define N_EDGES 800000
#define IN_DIM 256
#define OUT_DIM 64
#define H1 4
#define N_CLASSES 40
#define NEG_SLOPE 1.0f

typedef __attribute__((ext_vector_type(16))) __bf16 v16bf;
typedef __attribute__((ext_vector_type(8)))  __bf16 v8bf;
typedef __attribute__((ext_vector_type(8)))  float  v8f;

// ------------------------- monotone float<->uint for atomic max ------------
__device__ __forceinline__ unsigned f2mono(float f) {
  unsigned u = __float_as_uint(f);
  return (u & 0x80000000u) ? ~u : (u | 0x80000000u);
}
__device__ __forceinline__ float mono2f(unsigned m) {
  unsigned u = (m & 0x80000000u) ? (m & 0x7FFFFFFFu) : ~m;
  return __uint_as_float(u);
}

// ------------------------- f32 -> bf16 hi/lo split (row-major) -------------
__global__ void split_f32_bf16(const float* __restrict__ x,
                               __bf16* __restrict__ hi, __bf16* __restrict__ lo,
                               long long n) {
  const long long t = (long long)blockIdx.x * blockDim.x + threadIdx.x;
  if (t >= n) return;
  const float v = x[t];
  const __bf16 h = (__bf16)v;
  hi[t] = h;
  lo[t] = (__bf16)(v - (float)h);
}

// ------------------------- pack B into fragment-major bf16 hi/lo -----------
// Fragment f = kt*Ntiles + nt covers K rows [kt*32, kt*32+32), N cols
// [nt*16, nt*16+16) of W (K x N row-major f32). Per CDNA5 B layout
// (32x16 bf16): lane L -> N = nt*16 + (L&15); element e -> K = kt*32 +
// 8*(L>=16) + e (e<8), kt*32 + 16 + 8*(L>=16) + (e-8) (e>=8). Stored as 16 consecutive bf16 per lane (32B vector load).
__global__ void pack_b_frags(const float* __restrict__ W,
                             __bf16* __restrict__ bhi, __bf16* __restrict__ blo,
                             int K, int N, int Ntiles) {
  const int t = blockIdx.x * blockDim.x + threadIdx.x;
  const int lane = t & 31;
  const int f = t >> 5;
  const int nfrags = (K >> 5) * Ntiles;
  if (f >= nfrags) return;
  const int kt = f / Ntiles;
  const int nt = f - kt * Ntiles;
  const int n = nt * 16 + (lane & 15);
  const int kb = kt * 32 + ((lane >> 4) << 3);
  const bool ok = (n < N);
  v16bf h, l;
#pragma unroll
  for (int e = 0; e < 16; ++e) {
    const float x = ok ? W[(size_t)(kb + e + ((e >> 3) << 3)) * N + n] : 0.0f;
    const __bf16 xh = (__bf16)x;
    h[e] = xh;
    l[e] = (__bf16)(x - (float)xh);
  }
  const size_t off = ((size_t)f * 32 + lane) * 16;
  *(v16bf*)(bhi + off) = h;
  *(v16bf*)(blo + off) = l;
}

// ------------------------- A fragment load (row-major bf16) ----------------
// lane L: M = L&15; elems 0..7 -> K = kb + 8*(L>=16) + e, elems 8..15 ->
// K = kb + 16 + 8*(L>=16) + (e-8)  => two 16B loads, free shuffle.
__device__ __forceinline__ v16bf load_a16(const __bf16* p) {
  return __builtin_shufflevector(
      *(const v8bf*)p, *(const v8bf*)(p + 16),
      0, 1, 2, 3, 4, 5, 6, 7, 8, 9, 10, 11, 12, 13, 14, 15);
}

// ------------------------- WMMA GEMM (pre-split operands) ------------------
// One wave per 32x16 C block (2 M-tiles x 1 N-tile). K multiple of 32,
// M multiple of 16 (odd tile count handled by clamping the second M-tile
// -- duplicate compute, guarded store -- so all branches stay wave-uniform
// and EXEC is all-ones around every WMMA).
__global__ void gemm_wmma_packed(const __bf16* __restrict__ Ahi,
                                 const __bf16* __restrict__ Alo,
                                 const __bf16* __restrict__ Bhi,
                                 const __bf16* __restrict__ Blo,
                                 float* __restrict__ C,
                                 int M, int N, int K, int Ntiles) {
  const int lane = threadIdx.x & 31;
  const int wave = threadIdx.x >> 5;
  const int tile = blockIdx.x * (blockDim.x >> 5) + wave;
  const int Mtiles = M >> 4;
  const int Pairs = (Mtiles + 1) >> 1;
  if (tile >= Pairs * Ntiles) return;  // whole-wave exit
  const int mp = tile / Ntiles;
  const int ntile = tile - mp * Ntiles;
  const int m0 = mp * 2;
  const int m1 = (m0 + 1 < Mtiles) ? (m0 + 1) : m0;  // clamped tail

  const size_t aoff = (size_t)(lane & 15) * K + ((lane >> 4) << 3);
  const __bf16* pah0 = Ahi + (size_t)m0 * 16 * K + aoff;
  const __bf16* pal0 = Alo + (size_t)m0 * 16 * K + aoff;
  const __bf16* pah1 = Ahi + (size_t)m1 * 16 * K + aoff;
  const __bf16* pal1 = Alo + (size_t)m1 * 16 * K + aoff;
  const __bf16* pbh = Bhi + ((size_t)ntile * 32 + lane) * 16;
  const __bf16* pbl = Blo + ((size_t)ntile * 32 + lane) * 16;
  const size_t bstep = (size_t)Ntiles * 32 * 16;  // bf16 per K-step

  v8f c00 = {}, c01 = {}, c02 = {};
  v8f c10 = {}, c11 = {}, c12 = {};
  for (int k = 0; k < K; k += 32) {
    const v16bf bh = *(const v16bf*)pbh;
    const v16bf bl = *(const v16bf*)pbl;
    pbh += bstep;
    pbl += bstep;
    const v16bf a0h = load_a16(pah0 + k);
    const v16bf a0l = load_a16(pal0 + k);
    const v16bf a1h = load_a16(pah1 + k);
    const v16bf a1l = load_a16(pal1 + k);
    // 6 independent accumulator chains per K-step.
    c00 = __builtin_amdgcn_wmma_f32_16x16x32_bf16(false, a0h, false, bh,
                                                  (short)0, c00, false, false);
    c10 = __builtin_amdgcn_wmma_f32_16x16x32_bf16(false, a1h, false, bh,
                                                  (short)0, c10, false, false);
    c01 = __builtin_amdgcn_wmma_f32_16x16x32_bf16(false, a0h, false, bl,
                                                  (short)0, c01, false, false);
    c11 = __builtin_amdgcn_wmma_f32_16x16x32_bf16(false, a1h, false, bl,
                                                  (short)0, c11, false, false);
    c02 = __builtin_amdgcn_wmma_f32_16x16x32_bf16(false, a0l, false, bh,
                                                  (short)0, c02, false, false);
    c12 = __builtin_amdgcn_wmma_f32_16x16x32_bf16(false, a1l, false, bh,
                                                  (short)0, c12, false, false);
  }

  // C layout: VGPR r -> M = r + 8*(lane>=16); N = lane&15.
  const int n = ntile * 16 + (lane & 15);
  if (n < N) {
    const int rb = (lane >> 4) << 3;
#pragma unroll
    for (int r = 0; r < 8; ++r)
      C[(size_t)(m0 * 16 + rb + r) * N + n] = c00[r] + c01[r] + c02[r];
    if (m0 + 1 < Mtiles) {
#pragma unroll
      for (int r = 0; r < 8; ++r)
        C[(size_t)(m1 * 16 + rb + r) * N + n] = c10[r] + c11[r] + c12[r];
    }
  }
}

// ------------------------- attention scores el/er per (node, head) ---------
__global__ void attn_scores(const float* __restrict__ h,
                            const float* __restrict__ al,
                            const float* __restrict__ ar,
                            float* __restrict__ el, float* __restrict__ er,
                            int Nn, int H, int D) {
  const int t = blockIdx.x * blockDim.x + threadIdx.x;
  if (t >= Nn * H) return;
  const int node = t / H, hh = t - node * H;
  const float* hp = h + ((size_t)node * H + hh) * D;
  float sl = 0.f, sr = 0.f;
  for (int d = 0; d < D; ++d) {
    const float v = hp[d];
    sl += v * al[hh * D + d];
    sr += v * ar[hh * D + d];
  }
  el[t] = sl;
  er[t] = sr;
}

// ------------------------- edge softmax pass 1: segment max ----------------
__global__ void edge_max(const int* __restrict__ src, const int* __restrict__ dst,
                         const float* __restrict__ el, const float* __restrict__ er,
                         unsigned* __restrict__ emax, int E, int H) {
  const int t = blockIdx.x * blockDim.x + threadIdx.x;
  if (t >= E * H) return;
  const int e = t / H, hh = t - e * H;
  const int s = src[e], d = dst[e];
  float x = el[s * H + hh] + er[d * H + hh];
  x = (x >= 0.f) ? x : NEG_SLOPE * x;
  atomicMax(&emax[d * H + hh], f2mono(x));
}

// ------------------------- edge softmax pass 2: exp + segment sum ----------
__global__ void edge_exp(const int* __restrict__ src, const int* __restrict__ dst,
                         const float* __restrict__ el, const float* __restrict__ er,
                         const unsigned* __restrict__ emax,
                         float* __restrict__ aE, float* __restrict__ denom,
                         int E, int H) {
  const int t = blockIdx.x * blockDim.x + threadIdx.x;
  if (t >= E * H) return;
  const int e = t / H, hh = t - e * H;
  const int s = src[e], d = dst[e];
  float x = el[s * H + hh] + er[d * H + hh];
  x = (x >= 0.f) ? x : NEG_SLOPE * x;
  const float m = mono2f(emax[d * H + hh]);
  const float a = expf(x - m);
  aE[t] = a;
  atomicAdd(&denom[d * H + hh], a);
}

// ------------------------- edge aggregation: out[dst] += w * h[src] --------
// One thread per (edge, head, 4-dim chunk): 128-bit gather + 4 f32 atomics.
__global__ void edge_aggregate(const int* __restrict__ src, const int* __restrict__ dst,
                               const float* __restrict__ aE, const float* __restrict__ denom,
                               const float* __restrict__ h, float* __restrict__ out,
                               int E, int H, int D) {
  const int chunks = (H * D) >> 2;
  const long long t = (long long)blockIdx.x * blockDim.x + threadIdx.x;
  if (t >= (long long)E * chunks) return;
  const int e = (int)(t / chunks);
  const int r = (int)(t - (long long)e * chunks);
  const int dq = D >> 2;
  const int hh = r / dq;
  const int d0 = (r - hh * dq) << 2;
  const int s = src[e], d = dst[e];
  const float w = aE[(size_t)e * H + hh] / fmaxf(denom[d * H + hh], 1e-9f);
  const float4 v = *(const float4*)(h + ((size_t)s * H + hh) * D + d0);
  float* op = out + ((size_t)d * H + hh) * D + d0;
  atomicAdd(op + 0, v.x * w);
  atomicAdd(op + 1, v.y * w);
  atomicAdd(op + 2, v.z * w);
  atomicAdd(op + 3, v.w * w);
}

// ------------------------- bias + ELU (f32 out) ----------------------------
__global__ void bias_elu(const float* __restrict__ acc, const float* __restrict__ bias,
                         float* __restrict__ out, int Nn, int C) {
  const int t = blockIdx.x * blockDim.x + threadIdx.x;
  if (t >= Nn * C) return;
  const int c = t % C;
  const float x = acc[t] + bias[c];
  out[t] = (x > 0.f) ? x : (expf(x) - 1.f);
}

// ------------------------- bias + ELU -> bf16 hi/lo (feeds next GEMM) ------
__global__ void bias_elu_split(const float* __restrict__ acc, const float* __restrict__ bias,
                               __bf16* __restrict__ hi, __bf16* __restrict__ lo,
                               int Nn, int C) {
  const int t = blockIdx.x * blockDim.x + threadIdx.x;
  if (t >= Nn * C) return;
  const int c = t % C;
  float x = acc[t] + bias[c];
  x = (x > 0.f) ? x : (expf(x) - 1.f);
  const __bf16 h = (__bf16)x;
  hi[t] = h;
  lo[t] = (__bf16)(x - (float)h);
}

// ---------------------------------------------------------------------------
extern "C" void kernel_launch(void* const* d_in, const int* in_sizes, int n_in,
                              void* d_out, int out_size, void* d_ws, size_t ws_size,
                              hipStream_t stream) {
  (void)in_sizes; (void)n_in; (void)out_size; (void)ws_size;
  const float* feat = (const float*)d_in[0];
  const float* W1   = (const float*)d_in[1];
  const float* al1  = (const float*)d_in[2];
  const float* ar1  = (const float*)d_in[3];
  const float* b1   = (const float*)d_in[4];
  const float* W2   = (const float*)d_in[5];
  const float* al2  = (const float*)d_in[6];
  const float* ar2  = (const float*)d_in[7];
  const float* b2   = (const float*)d_in[8];
  const int*   src  = (const int*)d_in[9];
  const int*   dst  = (const int*)d_in[10];
  float* out = (float*)d_out;

  // -------- workspace carve-up --------
  const size_t F1 = (size_t)H1 * OUT_DIM;          // 256
  char* p = (char*)d_ws;
  auto take = [&](size_t bytes) { void* q = p; p += ((bytes + 255) & ~(size_t)255); return q; };
  float*    h1    = (float*)take((size_t)N_NODES * F1 * 4);          // 51.2 MB
  float*    x1a   = (float*)take((size_t)N_NODES * F1 * 4);          // 51.2 MB (agg accum)
  __bf16*   Ahi   = (__bf16*)take((size_t)N_NODES * F1 * 2);         // 25.6 MB (feat, then x1)
  __bf16*   Alo   = (__bf16*)take((size_t)N_NODES * F1 * 2);         // 25.6 MB
  __bf16*   W1hi  = (__bf16*)take((size_t)IN_DIM * F1 * 2);          // 128 KB (frag-major)
  __bf16*   W1lo  = (__bf16*)take((size_t)IN_DIM * F1 * 2);
  __bf16*   W2hi  = (__bf16*)take((size_t)F1 * 48 * 2);              // 24 KB (padded 48)
  __bf16*   W2lo  = (__bf16*)take((size_t)F1 * 48 * 2);
  float*    el1   = (float*)take((size_t)N_NODES * H1 * 4);
  float*    er1   = (float*)take((size_t)N_NODES * H1 * 4);
  unsigned* emax1 = (unsigned*)take((size_t)N_NODES * H1 * 4);
  float*    den1  = (float*)take((size_t)N_NODES * H1 * 4);
  float*    aE1   = (float*)take((size_t)N_EDGES * H1 * 4);          // 12.8 MB
  float*    h2    = (float*)take((size_t)N_NODES * N_CLASSES * 4);
  float*    acc2  = (float*)take((size_t)N_NODES * N_CLASSES * 4);
  float*    el2   = (float*)take((size_t)N_NODES * 4);
  float*    er2   = (float*)take((size_t)N_NODES * 4);
  unsigned* emax2 = (unsigned*)take((size_t)N_NODES * 4);
  float*    den2  = (float*)take((size_t)N_NODES * 4);
  float*    aE2   = (float*)take((size_t)N_EDGES * 4);

  // -------- zero accumulators (graph-capturable) --------
  hipMemsetAsync(x1a,   0, (size_t)N_NODES * F1 * 4, stream);
  hipMemsetAsync(emax1, 0, (size_t)N_NODES * H1 * 4, stream);  // mono 0 == -inf
  hipMemsetAsync(den1,  0, (size_t)N_NODES * H1 * 4, stream);
  hipMemsetAsync(acc2,  0, (size_t)N_NODES * N_CLASSES * 4, stream);
  hipMemsetAsync(emax2, 0, (size_t)N_NODES * 4, stream);
  hipMemsetAsync(den2,  0, (size_t)N_NODES * 4, stream);

  const int TB = 256;
  auto blocks = [](long long n, int tb) { return (unsigned)((n + tb - 1) / tb); };

  // -------- operand prep --------
  split_f32_bf16<<<blocks((long long)N_NODES * F1, TB), TB, 0, stream>>>(
      feat, Ahi, Alo, (long long)N_NODES * F1);
  pack_b_frags<<<blocks((long long)(IN_DIM / 32) * 16 * 32, TB), TB, 0, stream>>>(
      W1, W1hi, W1lo, IN_DIM, (int)F1, 16);
  pack_b_frags<<<blocks((long long)((int)F1 / 32) * 3 * 32, TB), TB, 0, stream>>>(
      W2, W2hi, W2lo, (int)F1, N_CLASSES, 3);

  const int Mtiles = N_NODES / 16;            // 3125
  const int Pairs  = (Mtiles + 1) / 2;        // 1563

  // ================= layer 1 =================
  gemm_wmma_packed<<<blocks((long long)Pairs * 16, 8), 256, 0, stream>>>(
      Ahi, Alo, W1hi, W1lo, h1, N_NODES, (int)F1, IN_DIM, 16);
  attn_scores<<<blocks((long long)N_NODES * H1, TB), TB, 0, stream>>>(
      h1, al1, ar1, el1, er1, N_NODES, H1, OUT_DIM);
  edge_max<<<blocks((long long)N_EDGES * H1, TB), TB, 0, stream>>>(
      src, dst, el1, er1, emax1, N_EDGES, H1);
  edge_exp<<<blocks((long long)N_EDGES * H1, TB), TB, 0, stream>>>(
      src, dst, el1, er1, emax1, aE1, den1, N_EDGES, H1);
  edge_aggregate<<<blocks((long long)N_EDGES * (H1 * OUT_DIM / 4), TB), TB, 0, stream>>>(
      src, dst, aE1, den1, h1, x1a, N_EDGES, H1, OUT_DIM);
  bias_elu_split<<<blocks((long long)N_NODES * F1, TB), TB, 0, stream>>>(
      x1a, b1, Ahi, Alo, N_NODES, (int)F1);   // x1 -> bf16 hi/lo (reuses feat bufs)

  // ================= layer 2 =================
  gemm_wmma_packed<<<blocks((long long)Pairs * 3, 8), 256, 0, stream>>>(
      Ahi, Alo, W2hi, W2lo, h2, N_NODES, N_CLASSES, (int)F1, 3);
  attn_scores<<<blocks((long long)N_NODES, TB), TB, 0, stream>>>(
      h2, al2, ar2, el2, er2, N_NODES, 1, N_CLASSES);
  edge_max<<<blocks((long long)N_EDGES, TB), TB, 0, stream>>>(
      src, dst, el2, er2, emax2, N_EDGES, 1);
  edge_exp<<<blocks((long long)N_EDGES, TB), TB, 0, stream>>>(
      src, dst, el2, er2, emax2, aE2, den2, N_EDGES, 1);
  edge_aggregate<<<blocks((long long)N_EDGES * (N_CLASSES / 4), TB), TB, 0, stream>>>(
      src, dst, aE2, den2, h2, acc2, N_EDGES, 1, N_CLASSES);
  bias_elu<<<blocks((long long)N_NODES * N_CLASSES, TB), TB, 0, stream>>>(
      acc2, b2, out, N_NODES, N_CLASSES);
}
